// HunyuanVideoSingleTransformerBlock_81664508166185
// MI455X (gfx1250) — compile-verified, matrix operand order checked by reading, not run on hardware
//
#include <hip/hip_runtime.h>
#include <cstdint>
#include <cstddef>

typedef __attribute__((ext_vector_type(16))) _Float16 v16h;
typedef __attribute__((ext_vector_type(8)))  float    v8f;

#define HIDDEN   3072
#define HEADS    24
#define HEAD_DIM 128
#define MLP_DIM  12288
#define S_TOT    2304
#define S_IMG    2048

// ---------------------------------------------------------------------------
// WMMA helpers (CDNA5 gfx1250, wave32). Layouts per CDNA5 ISA 7.12.2:
//  A (16x32 f16): lane L -> M=L&15; VGPR v, half-pair -> K per table
//  B (32x16 f16): lane L -> N=L&15; same K mapping (K pairs contiguous)
//  C/D (16x16 f32): lane L -> N=L&15; VGPR t -> M = t + 8*(L>>4)
// ---------------------------------------------------------------------------
__device__ __forceinline__ v8f wmma_f16(v16h a, v16h b, v8f c) {
  return __builtin_amdgcn_wmma_f32_16x16x32_f16(false, a, false, b, (short)0, c,
                                                false, false);
}

// p = pointer to this lane's row (K-contiguous), g = lane>>4
__device__ __forceinline__ v16h load_frag(const _Float16* p, int g) {
  v16h f;
#pragma unroll
  for (int u = 0; u < 8; ++u) {
    int kb = ((u < 4) ? (2 * u) : (16 + 2 * (u - 4))) + 8 * g;
    f[2 * u]     = p[kb];
    f[2 * u + 1] = p[kb + 1];
  }
  return f;
}

__device__ __forceinline__ float gelu_tanh(float x) {
  return 0.5f * x * (1.0f + tanhf(0.7978845608028654f * (x + 0.044715f * x * x * x)));
}

// ---------------------------------------------------------------------------
// Kernel 1: emb = silu(temb) @ w_norm + b_norm  (GEMV, memory-bound, fp32)
// grid 36 x 256 threads, one output column per thread
// ---------------------------------------------------------------------------
__global__ __launch_bounds__(256) void emb_gemv_kern(
    const float* __restrict__ temb, const float* __restrict__ w_norm,
    const float* __restrict__ b_norm, float* __restrict__ emb) {
  __shared__ float s[HIDDEN];
  for (int i = threadIdx.x; i < HIDDEN; i += 256) {
    float t = temb[i];
    s[i] = t / (1.0f + __expf(-t));  // silu
  }
  __syncthreads();
  int n = blockIdx.x * 256 + threadIdx.x;  // 0..9215
  float acc = b_norm[n];
  for (int r2 = 0; r2 < HIDDEN; ++r2)
    acc += s[r2] * w_norm[(size_t)r2 * (3 * HIDDEN) + n];
  emb[n] = acc;
}

// ---------------------------------------------------------------------------
// Kernel 2: nx = layernorm(concat(hs,ehs)) * (1+scale) + shift -> f16
// one block per token
// ---------------------------------------------------------------------------
__global__ __launch_bounds__(256) void ln_mod_kern(
    const float* __restrict__ hs, const float* __restrict__ ehs,
    const float* __restrict__ emb, _Float16* __restrict__ nxh) {
  int t = blockIdx.x;
  const float* x = (t < S_IMG) ? (hs + (size_t)t * HIDDEN)
                               : (ehs + (size_t)(t - S_IMG) * HIDDEN);
  __shared__ float red[256];
  float vals[12];
  float sum = 0.0f;
#pragma unroll
  for (int i = 0; i < 12; ++i) {
    vals[i] = x[threadIdx.x + i * 256];
    sum += vals[i];
  }
  red[threadIdx.x] = sum;
  __syncthreads();
  for (int s2 = 128; s2 > 0; s2 >>= 1) {
    if (threadIdx.x < s2) red[threadIdx.x] += red[threadIdx.x + s2];
    __syncthreads();
  }
  float mu = red[0] * (1.0f / HIDDEN);
  __syncthreads();
  float sq = 0.0f;
#pragma unroll
  for (int i = 0; i < 12; ++i) {
    float d = vals[i] - mu;
    sq += d * d;
  }
  red[threadIdx.x] = sq;
  __syncthreads();
  for (int s2 = 128; s2 > 0; s2 >>= 1) {
    if (threadIdx.x < s2) red[threadIdx.x] += red[threadIdx.x + s2];
    __syncthreads();
  }
  float inv = rsqrtf(red[0] * (1.0f / HIDDEN) + 1e-6f);
#pragma unroll
  for (int i = 0; i < 12; ++i) {
    int c = threadIdx.x + i * 256;
    float nv = (vals[i] - mu) * inv;
    nv = nv * (1.0f + emb[HIDDEN + c]) + emb[c];  // scale, shift
    nxh[(size_t)t * HIDDEN + c] = (_Float16)nv;
  }
}

// ---------------------------------------------------------------------------
// Kernel 3: tiled WMMA GEMM  C[M,N] = act(A_f16[M,K] @ W_f32[K,N] + bias)
// BM=128 BN=64 BK=32, 8 waves, each wave -> 32x32 (2x2 WMMA subtiles).
// MODE 0: +bias -> f16 out
// MODE 1: gelu(+bias) -> f16 out
// MODE 2: A = concat(attn | mlp) at ksplit; out = gate*( +bias ) + residual -> f32
// ---------------------------------------------------------------------------
template <int MODE>
__global__ __launch_bounds__(256) void gemm_f16_kern(
    const _Float16* __restrict__ A, int lda, const _Float16* __restrict__ A2,
    int lda2, int ksplit, const float* __restrict__ W,
    const float* __restrict__ bias, _Float16* __restrict__ outH,
    float* __restrict__ outF, const float* __restrict__ gate,
    const float* __restrict__ res0, const float* __restrict__ res1, int M,
    int N, int K) {
  __shared__ _Float16 sA[128][40];  // stride 40 halfs = 80B (16B multiple)
  __shared__ _Float16 sB[64][40];   // W tile transposed: [n][k]
  const int tid = threadIdx.x;
  const int wave = tid >> 5, lane = tid & 31;
  const int g = lane >> 4, r = lane & 15;
  const int wm = wave & 3, wn = wave >> 2;  // 4 x 2 wave grid
  const int mbase = blockIdx.y * 128;
  const int nbase = blockIdx.x * 64;

  v8f acc[2][2];
  {
    v8f z = {0, 0, 0, 0, 0, 0, 0, 0};
#pragma unroll
    for (int i = 0; i < 2; ++i)
#pragma unroll
      for (int j = 0; j < 2; ++j) acc[i][j] = z;
  }

  const int arow = tid >> 1;          // 0..127
  const int acol = (tid & 1) * 16;    // 0 or 16
  const int bk = tid >> 3;            // 0..31
  const int bn = (tid & 7) * 8;       // 0..56

  for (int k0 = 0; k0 < K; k0 += 32) {
    __syncthreads();
    // ---- stage A tile (f16, 128x32) ----
    {
      int gk = k0 + acol;
      const _Float16* src;
      if (MODE == 2 && gk >= ksplit)
        src = A2 + (size_t)(mbase + arow) * lda2 + (gk - ksplit);
      else
        src = A + (size_t)(mbase + arow) * lda + gk;
      const uint4* s4 = (const uint4*)src;
      uint4 d0 = s4[0];
      uint4 d1 = s4[1];
      *(uint4*)(&sA[arow][acol]) = d0;
      *(uint4*)(&sA[arow][acol + 8]) = d1;
      if (k0 + 32 < K) __builtin_prefetch((const void*)(src + 32), 0, 1);
    }
    // ---- stage B tile (f32 -> f16, transposed to [n][k]) ----
    {
      const float* wsrc = W + (size_t)(k0 + bk) * N + nbase + bn;
      float4 w0 = ((const float4*)wsrc)[0];
      float4 w1 = ((const float4*)wsrc)[1];
      sB[bn + 0][bk] = (_Float16)w0.x;
      sB[bn + 1][bk] = (_Float16)w0.y;
      sB[bn + 2][bk] = (_Float16)w0.z;
      sB[bn + 3][bk] = (_Float16)w0.w;
      sB[bn + 4][bk] = (_Float16)w1.x;
      sB[bn + 5][bk] = (_Float16)w1.y;
      sB[bn + 6][bk] = (_Float16)w1.z;
      sB[bn + 7][bk] = (_Float16)w1.w;
      if (k0 + 32 < K)
        __builtin_prefetch((const void*)(wsrc + (size_t)32 * N), 0, 1);
    }
    __syncthreads();
    // ---- compute 2x2 WMMA subtiles ----
    v16h a0 = load_frag(&sA[wm * 32 + 0 + r][0], g);
    v16h a1 = load_frag(&sA[wm * 32 + 16 + r][0], g);
    v16h b0 = load_frag(&sB[wn * 32 + 0 + r][0], g);
    v16h b1 = load_frag(&sB[wn * 32 + 16 + r][0], g);
    acc[0][0] = wmma_f16(a0, b0, acc[0][0]);
    acc[0][1] = wmma_f16(a0, b1, acc[0][1]);
    acc[1][0] = wmma_f16(a1, b0, acc[1][0]);
    acc[1][1] = wmma_f16(a1, b1, acc[1][1]);
  }

  // ---- epilogue ----
#pragma unroll
  for (int i = 0; i < 2; ++i) {
#pragma unroll
    for (int j = 0; j < 2; ++j) {
      int col = nbase + wn * 32 + j * 16 + r;
      float b = bias[col];
#pragma unroll
      for (int t = 0; t < 8; ++t) {
        int row = mbase + wm * 32 + i * 16 + t + 8 * g;
        float val = acc[i][j][t] + b;
        if (MODE == 1) val = gelu_tanh(val);
        if (MODE != 2) {
          outH[(size_t)row * N + col] = (_Float16)val;
        } else {
          float resv = (row < S_IMG)
                           ? res0[(size_t)row * HIDDEN + col]
                           : res1[(size_t)(row - S_IMG) * HIDDEN + col];
          outF[(size_t)row * N + col] = gate[col] * val + resv;
        }
      }
    }
  }
}

// ---------------------------------------------------------------------------
// Kernel 4: per-(token, head) RMSNorm on q,k + RoPE on image tokens, in place
// grid (2304, 24), block 128 (thread = head-dim element)
// ---------------------------------------------------------------------------
__global__ __launch_bounds__(128) void qk_norm_rope_kern(
    _Float16* __restrict__ q, _Float16* __restrict__ k,
    const float* __restrict__ nqw, const float* __restrict__ nkw,
    const float* __restrict__ freqs) {
  int t = blockIdx.x, h = blockIdx.y, d = threadIdx.x;
  __shared__ float red[128];
#pragma unroll
  for (int which = 0; which < 2; ++which) {
    _Float16* base = (which ? k : q) + (size_t)t * HIDDEN + h * HEAD_DIM;
    float x = (float)base[d];
    red[d] = x * x;
    __syncthreads();
    for (int s2 = 64; s2 > 0; s2 >>= 1) {
      if (d < s2) red[d] += red[d + s2];
      __syncthreads();
    }
    float inv = rsqrtf(red[0] * (1.0f / HEAD_DIM) + 1e-6f);
    float w = which ? nkw[d] : nqw[d];
    float val = x * inv * w;
    float out = val;
    if (t < S_IMG) {
      float partner = __shfl_xor(val, 1, 32);  // d^1 lives in same wave32
      float rot = (d & 1) ? partner : -partner;
      float c = freqs[(size_t)t * 256 + d];
      float s = freqs[(size_t)t * 256 + 128 + d];
      out = val * c + rot * s;
    }
    base[d] = (_Float16)out;
    __syncthreads();
  }
}

// ---------------------------------------------------------------------------
// Kernel 5: flash attention.  grid (36 q-tiles of 64, 24 heads), block 128.
// Wave w owns 16 q-rows; online softmax stats per row held in the 16-lane
// half-wave that owns the row (C-layout: VGPR t -> row t+8g, lane -> col).
// ---------------------------------------------------------------------------
__global__ __launch_bounds__(128) void attn_fa_kern(
    const _Float16* __restrict__ q, const _Float16* __restrict__ k,
    const _Float16* __restrict__ v, _Float16* __restrict__ o) {
  const int h = blockIdx.y;
  const int qbase = blockIdx.x * 64;
  const int tid = threadIdx.x, wave = tid >> 5, lane = tid & 31;
  const int g = lane >> 4, r = lane & 15;
  const int qrow = qbase + wave * 16;

  __shared__ _Float16 sK[32][136];    // [token][d], stride 272B
  __shared__ _Float16 sV[128][36];    // [d][token] (transposed for B frags)
  __shared__ _Float16 sP[4][16][40];  // per-wave P bounce (C-layout -> A-layout)

  // Q fragments for this wave's 16 rows (K = head dim, 4 chunks of 32)
  v16h aq[4];
  {
    const _Float16* qp = q + (size_t)(qrow + r) * HIDDEN + h * HEAD_DIM;
#pragma unroll
    for (int c = 0; c < 4; ++c) aq[c] = load_frag(qp + c * 32, g);
  }

  v8f o_acc[8];
  {
    v8f z = {0, 0, 0, 0, 0, 0, 0, 0};
#pragma unroll
    for (int t = 0; t < 8; ++t) o_acc[t] = z;
  }
  float m_i[8], l_i[8];
#pragma unroll
  for (int i = 0; i < 8; ++i) {
    m_i[i] = -1e30f;
    l_i[i] = 0.0f;
  }
  const float scale = 0.08838834764831845f;  // 1/sqrt(128)

  for (int j0 = 0; j0 < S_TOT; j0 += 32) {
    __syncthreads();
    // ---- cooperative stage of K,V tile (32 tokens x 128) ----
    {
      int tok = tid >> 2;            // 0..31
      int d0 = (tid & 3) * 32;       // 0,32,64,96
      const uint4* ksrc =
          (const uint4*)(k + (size_t)(j0 + tok) * HIDDEN + h * HEAD_DIM + d0);
#pragma unroll
      for (int q4 = 0; q4 < 4; ++q4)
        *(uint4*)(&sK[tok][d0 + q4 * 8]) = ksrc[q4];
      const uint4* vsrc =
          (const uint4*)(v + (size_t)(j0 + tok) * HIDDEN + h * HEAD_DIM + d0);
#pragma unroll
      for (int q4 = 0; q4 < 4; ++q4) {
        uint4 pkt = vsrc[q4];
        const _Float16* hp = (const _Float16*)&pkt;
#pragma unroll
        for (int e = 0; e < 8; ++e) sV[d0 + q4 * 8 + e][tok] = hp[e];
      }
    }
    __syncthreads();

    // ---- S = Q @ K^T for 16x32 tile (two 16x16 WMMA accum chains) ----
    v8f s0 = {0, 0, 0, 0, 0, 0, 0, 0};
    v8f s1 = {0, 0, 0, 0, 0, 0, 0, 0};
#pragma unroll
    for (int c = 0; c < 4; ++c) {
      v16h b0 = load_frag(&sK[r][c * 32], g);
      v16h b1 = load_frag(&sK[16 + r][c * 32], g);
      s0 = wmma_f16(aq[c], b0, s0);
      s1 = wmma_f16(aq[c], b1, s1);
    }

    // ---- online softmax (row = VGPR i + 8g, reduce over 16-lane group) ----
#pragma unroll
    for (int i = 0; i < 8; ++i) {
      float x0 = s0[i] * scale, x1 = s1[i] * scale;
      float mx = fmaxf(x0, x1);
      mx = fmaxf(mx, __shfl_xor(mx, 1, 32));
      mx = fmaxf(mx, __shfl_xor(mx, 2, 32));
      mx = fmaxf(mx, __shfl_xor(mx, 4, 32));
      mx = fmaxf(mx, __shfl_xor(mx, 8, 32));
      float nm = fmaxf(m_i[i], mx);
      float corr = __expf(m_i[i] - nm);
      float p0 = __expf(x0 - nm), p1 = __expf(x1 - nm);
      float rs = p0 + p1;
      rs += __shfl_xor(rs, 1, 32);
      rs += __shfl_xor(rs, 2, 32);
      rs += __shfl_xor(rs, 4, 32);
      rs += __shfl_xor(rs, 8, 32);
      l_i[i] = l_i[i] * corr + rs;
      m_i[i] = nm;
#pragma unroll
      for (int t = 0; t < 8; ++t) o_acc[t][i] *= corr;
      // C-layout -> LDS (row-major) for A-layout reload (same-wave DS in-order)
      sP[wave][i + 8 * g][r] = (_Float16)p0;
      sP[wave][i + 8 * g][16 + r] = (_Float16)p1;
    }

    // ---- O += P(16x32) @ V(32x128) ----
    v16h ap = load_frag(&sP[wave][r][0], g);
#pragma unroll
    for (int t = 0; t < 8; ++t) {
      v16h bv = load_frag(&sV[t * 16 + r][0], g);
      o_acc[t] = wmma_f16(ap, bv, o_acc[t]);
    }
  }

  // ---- normalize + store attn (f16) ----
#pragma unroll
  for (int i = 0; i < 8; ++i) {
    float inv = 1.0f / l_i[i];
    int row = qrow + i + 8 * g;
#pragma unroll
    for (int t = 0; t < 8; ++t)
      o[(size_t)row * HIDDEN + h * HEAD_DIM + t * 16 + r] =
          (_Float16)(o_acc[t][i] * inv);
  }
}

// ---------------------------------------------------------------------------
// Host launch
// ---------------------------------------------------------------------------
extern "C" void kernel_launch(void* const* d_in, const int* in_sizes, int n_in,
                              void* d_out, int out_size, void* d_ws,
                              size_t ws_size, hipStream_t stream) {
  const float* hs     = (const float*)d_in[0];   // [1,2048,3072]
  const float* ehs    = (const float*)d_in[1];   // [1,256,3072]
  const float* temb   = (const float*)d_in[2];   // [1,3072]
  const float* freqs  = (const float*)d_in[3];   // [2048,256]
  const float* w_norm = (const float*)d_in[4];   // [3072,9216]
  const float* b_norm = (const float*)d_in[5];
  const float* wq     = (const float*)d_in[6];
  const float* bq     = (const float*)d_in[7];
  const float* wk     = (const float*)d_in[8];
  const float* bk     = (const float*)d_in[9];
  const float* wv     = (const float*)d_in[10];
  const float* bv     = (const float*)d_in[11];
  const float* nqw    = (const float*)d_in[12];
  const float* nkw    = (const float*)d_in[13];
  const float* w_mlp  = (const float*)d_in[14];  // [3072,12288]
  const float* b_mlp  = (const float*)d_in[15];
  const float* w_out  = (const float*)d_in[16];  // [15360,3072]
  const float* b_out  = (const float*)d_in[17];
  float* outF = (float*)d_out;                   // [2304,3072] fp32

  // workspace carve-up (bytes)
  char* ws = (char*)d_ws;
  constexpr size_t SZ_H = (size_t)S_TOT * HIDDEN * sizeof(_Float16);   // 14.16MB
  constexpr size_t SZ_M = (size_t)S_TOT * MLP_DIM * sizeof(_Float16);  // 56.6MB
  float*    emb   = (float*)(ws + 0);                 // 9216 f32
  _Float16* nxh   = (_Float16*)(ws + (1 << 16));
  _Float16* qh    = (_Float16*)(ws + (1 << 16) + 1 * SZ_H);
  _Float16* kh    = (_Float16*)(ws + (1 << 16) + 2 * SZ_H);
  _Float16* vh    = (_Float16*)(ws + (1 << 16) + 3 * SZ_H);
  _Float16* mlph  = (_Float16*)(ws + (1 << 16) + 4 * SZ_H);
  _Float16* attnh = (_Float16*)(ws + (1 << 16) + 4 * SZ_H + SZ_M);
  (void)in_sizes; (void)n_in; (void)out_size; (void)ws_size;

  // 1. modulation embedding
  emb_gemv_kern<<<36, 256, 0, stream>>>(temb, w_norm, b_norm, emb);
  // 2. layernorm + modulate -> nx (f16)
  ln_mod_kern<<<S_TOT, 256, 0, stream>>>(hs, ehs, emb, nxh);
  // 3. MLP up-proj + gelu -> mlp (f16)
  gemm_f16_kern<1><<<dim3(MLP_DIM / 64, S_TOT / 128), 256, 0, stream>>>(
      nxh, HIDDEN, nullptr, 0, 0, w_mlp, b_mlp, mlph, nullptr, nullptr,
      nullptr, nullptr, S_TOT, MLP_DIM, HIDDEN);
  // 4. Q, K, V projections -> f16
  gemm_f16_kern<0><<<dim3(HIDDEN / 64, S_TOT / 128), 256, 0, stream>>>(
      nxh, HIDDEN, nullptr, 0, 0, wq, bq, qh, nullptr, nullptr, nullptr,
      nullptr, S_TOT, HIDDEN, HIDDEN);
  gemm_f16_kern<0><<<dim3(HIDDEN / 64, S_TOT / 128), 256, 0, stream>>>(
      nxh, HIDDEN, nullptr, 0, 0, wk, bk, kh, nullptr, nullptr, nullptr,
      nullptr, S_TOT, HIDDEN, HIDDEN);
  gemm_f16_kern<0><<<dim3(HIDDEN / 64, S_TOT / 128), 256, 0, stream>>>(
      nxh, HIDDEN, nullptr, 0, 0, wv, bv, vh, nullptr, nullptr, nullptr,
      nullptr, S_TOT, HIDDEN, HIDDEN);
  // 5. RMSNorm(q,k) + RoPE on image tokens (in place)
  qk_norm_rope_kern<<<dim3(S_TOT, HEADS), 128, 0, stream>>>(qh, kh, nqw, nkw,
                                                            freqs);
  // 6. flash attention -> attn (f16)
  attn_fa_kern<<<dim3(S_TOT / 64, HEADS), 128, 0, stream>>>(qh, kh, vh, attnh);
  // 7. out-proj over concat(attn, mlp), gate + residual -> fp32 d_out
  gemm_f16_kern<2><<<dim3(HIDDEN / 64, S_TOT / 128), 256, 0, stream>>>(
      attnh, HIDDEN, mlph, MLP_DIM, HIDDEN, w_out, b_out, nullptr, outF,
      emb + 2 * HIDDEN, hs, ehs, S_TOT, HIDDEN, 15360);
}
